// TransformerModel_86380382257158
// MI455X (gfx1250) — compile-verified, matrix operand order checked by reading, not run on hardware
//
#include <hip/hip_runtime.h>
#include <cmath>

// ---------------------------------------------------------------------------
// Transformer encoder-decoder forward for MI455X (gfx1250), wave32 + WMMA.
// All GEMMs: bf16 WMMA 16x16x32, f32 accumulate. fp32->bf16 conversion happens
// in the LDS tile stage (no extra HBM traffic; weights are single-use so
// pre-conversion would be a net bandwidth loss).
// Tile fill: clause of 6 global_load_b128 -> cvt -> ds_store_b64 (loads issued
// before the barrier so they overlap the previous iteration's WMMA drain).
// ---------------------------------------------------------------------------

typedef __bf16 bf16_t;
typedef __attribute__((ext_vector_type(4)))  __bf16 bf16x4;
typedef __attribute__((ext_vector_type(16))) __bf16 v16bf;
typedef __attribute__((ext_vector_type(8)))  float  v8f;

#define TM 128
#define TN 64
#define TK 32

// C[M,N] = op( scale * A[M,K] x B[K,N] + bias[N] )
// B addressed through its transpose: B^T(n,k) = Bt[n*bsn + k*bsk].
// BCONTIG_K: true  -> bsk==1 (B^T rows contiguous: Q.K^T scores, logits)
//            false -> bsn==1 (B rows contiguous: row-major weights, P.V)
// Batched over blockIdx.z: base = (z/innerCnt)*Outer + (z%innerCnt)*Inner.
template <bool BCONTIG_K>
__global__ __launch_bounds__(256) void gemm_bf16_wmma(
    const float* __restrict__ A, int lda, int aOuter, int aInner,
    const float* __restrict__ Bt, int bsn, int bsk, int bOuter, int bInner,
    float* __restrict__ C, int ldc, int cOuter, int cInner,
    const float* __restrict__ bias, float scale, int relu,
    int M, int N, int K, int innerCnt)
{
    const int z  = blockIdx.z;
    const int zo = z / innerCnt;
    const int zi = z - zo * innerCnt;
    A  += zo * aOuter + zi * aInner;
    Bt += zo * bOuter + zi * bInner;
    C  += zo * cOuter + zi * cInner;

    __shared__ bf16_t As[TM][TK + 8];
    __shared__ bf16_t Bs[TN][TK + 8];

    const int tid  = threadIdx.x;
    const int lane = tid & 31;
    const int wave = tid >> 5;
    const int wm   = wave >> 1;   // 0..3 : 32-row strip
    const int wn   = wave & 1;    // 0..1 : 32-col strip
    const int l16  = lane & 15;
    const int hl   = lane >> 4;   // lane half (K-range selector)

    const int m0 = blockIdx.y * TM;
    const int n0 = blockIdx.x * TN;

    // ---- precompute per-thread tile pointers (walked by constant stride)
    const float4* ap[4];
    bf16x4*       adst[4];
    #pragma unroll
    for (int it = 0; it < 4; ++it) {
        int idx = tid + it * 256;            // 0..1023, one float4 each
        int r   = idx >> 3;                  // 8 float4 per 32-wide row
        int c4  = (idx & 7) * 4;
        ap[it]   = reinterpret_cast<const float4*>(&A[(m0 + r) * lda + c4]);
        adst[it] = reinterpret_cast<bf16x4*>(&As[r][c4]);
    }
    const float4* bp[2];
    bf16x4*       bdstv[2];                  // BCONTIG_K: vector LDS store
    bf16_t*       bdsts[2];                  // !BCONTIG_K: strided scatter
    #pragma unroll
    for (int it = 0; it < 2; ++it) {
        int idx = tid + it * 256;            // 0..511
        if (BCONTIG_K) {
            int n  = idx >> 3;
            int c4 = (idx & 7) * 4;
            bp[it]    = reinterpret_cast<const float4*>(&Bt[(n0 + n) * bsn + c4]);
            bdstv[it] = reinterpret_cast<bf16x4*>(&Bs[n][c4]);
        } else {
            int c  = idx >> 4;               // 32 K-columns
            int n4 = (idx & 15) * 4;         // 16 float4 across N
            bp[it]    = reinterpret_cast<const float4*>(&Bt[(n0 + n4) + c * bsk]);
            bdsts[it] = &Bs[n4][c];
        }
    }
    const int bstep = BCONTIG_K ? (TK / 4) : (TK / 4) * bsk;  // float4 units

    const v8f vzero = {0.f, 0.f, 0.f, 0.f, 0.f, 0.f, 0.f, 0.f};
    v8f acc[2][2];
    acc[0][0] = vzero; acc[0][1] = vzero; acc[1][0] = vzero; acc[1][1] = vzero;

    for (int k0 = 0; k0 < K; k0 += TK) {
        // ---- issue all 6 global b128 loads up front (single clause)
        float4 va[4], vb[2];
        #pragma unroll
        for (int it = 0; it < 4; ++it) { va[it] = *ap[it]; ap[it] += TK / 4; }
        #pragma unroll
        for (int it = 0; it < 2; ++it) { vb[it] = *bp[it]; bp[it] += bstep; }

        __syncthreads();   // previous iteration's fragments consumed

        // ---- convert fp32 -> bf16 and stage into LDS
        #pragma unroll
        for (int it = 0; it < 4; ++it) {
            bf16x4 t;
            t[0] = (bf16_t)va[it].x; t[1] = (bf16_t)va[it].y;
            t[2] = (bf16_t)va[it].z; t[3] = (bf16_t)va[it].w;
            *adst[it] = t;
        }
        #pragma unroll
        for (int it = 0; it < 2; ++it) {
            if (BCONTIG_K) {
                bf16x4 t;
                t[0] = (bf16_t)vb[it].x; t[1] = (bf16_t)vb[it].y;
                t[2] = (bf16_t)vb[it].z; t[3] = (bf16_t)vb[it].w;
                *bdstv[it] = t;
            } else {
                bf16_t* d = bdsts[it];
                d[0 * (TK + 8)] = (bf16_t)vb[it].x;
                d[1 * (TK + 8)] = (bf16_t)vb[it].y;
                d[2 * (TK + 8)] = (bf16_t)vb[it].z;
                d[3 * (TK + 8)] = (bf16_t)vb[it].w;
            }
        }
        __syncthreads();

        // ---- fragment assembly per CDNA5 16-bit layouts:
        // lanes 0-15 hold K 0-7 (regs 0-3) and 16-23 (regs 4-7);
        // lanes 16-31 hold K 8-15 and 24-31.  (lowers to ds_load_b128)
        v16bf afr[2], bfr[2];
        #pragma unroll
        for (int f = 0; f < 2; ++f) {
            int m = wm * 32 + f * 16 + l16;
            int n = wn * 32 + f * 16 + l16;
            v16bf av, bv;
            #pragma unroll
            for (int e = 0; e < 16; ++e) {
                int k = (e < 8) ? (hl * 8 + e) : (8 + hl * 8 + e);
                av[e] = As[m][k];
                bv[e] = Bs[n][k];
            }
            afr[f] = av; bfr[f] = bv;
        }

        #pragma unroll
        for (int fm = 0; fm < 2; ++fm)
            #pragma unroll
            for (int fn = 0; fn < 2; ++fn)
                acc[fm][fn] = __builtin_amdgcn_wmma_f32_16x16x32_bf16(
                    false, afr[fm], false, bfr[fn],
                    (short)0, acc[fm][fn], false, false);
    }

    // ---- store: C frag layout -> n = l16, m = hl*8 + reg
    #pragma unroll
    for (int fm = 0; fm < 2; ++fm) {
        #pragma unroll
        for (int fn = 0; fn < 2; ++fn) {
            int n = n0 + wn * 32 + fn * 16 + l16;
            float bv = bias ? bias[n] : 0.f;
            #pragma unroll
            for (int r = 0; r < 8; ++r) {
                int m = m0 + wm * 32 + fm * 16 + hl * 8 + r;
                float v = acc[fm][fn][r] * scale + bv;
                if (relu) v = fmaxf(v, 0.f);
                C[m * ldc + n] = v;
            }
        }
    }
}

// ---- masked softmax over rows of length ncols (=512); 2 cols per thread.
// blocked(col) = (causal && col>row_pos) || (ktok && ktok[b,col]==0); ref adds -1e9.
__global__ __launch_bounds__(256) void softmax_mask_k(
    float* __restrict__ sc, const int* __restrict__ ktok,
    int causal, int ncols, int rows_per_batch)
{
    __shared__ float red[256];
    const int row  = blockIdx.x;
    const int tid  = threadIdx.x;
    const int b    = row / rows_per_batch;
    const int srow = row % ncols;
    const int base = row * ncols;

    float x[2];
    #pragma unroll
    for (int j = 0; j < 2; ++j) {
        int col = tid + j * 256;
        float v = sc[base + col];
        bool blocked = (causal && col > srow) ||
                       (ktok && ktok[b * ncols + col] == 0);
        x[j] = blocked ? v - 1e9f : v;
    }
    red[tid] = fmaxf(x[0], x[1]); __syncthreads();
    for (int s = 128; s > 0; s >>= 1) {
        if (tid < s) red[tid] = fmaxf(red[tid], red[tid + s]);
        __syncthreads();
    }
    float mx = red[0]; __syncthreads();
    float e0 = __expf(x[0] - mx), e1 = __expf(x[1] - mx);
    red[tid] = e0 + e1; __syncthreads();
    for (int s = 128; s > 0; s >>= 1) {
        if (tid < s) red[tid] += red[tid + s];
        __syncthreads();
    }
    float inv = 1.f / red[0];
    sc[base + tid]       = e0 * inv;
    sc[base + tid + 256] = e1 * inv;
}

// ---- x = LayerNorm(x + y) * g + b, row length 512, 2 elems/thread
__global__ __launch_bounds__(256) void add_ln_k(
    float* __restrict__ x, const float* __restrict__ y,
    const float* __restrict__ g, const float* __restrict__ bta, int Dim)
{
    __shared__ float red[256];
    const int base = blockIdx.x * Dim;
    const int tid = threadIdx.x;
    float s0 = x[base + tid]       + y[base + tid];
    float s1 = x[base + tid + 256] + y[base + tid + 256];
    red[tid] = s0 + s1; __syncthreads();
    for (int s = 128; s > 0; s >>= 1) {
        if (tid < s) red[tid] += red[tid + s];
        __syncthreads();
    }
    float mean = red[0] / Dim; __syncthreads();
    float d0 = s0 - mean, d1 = s1 - mean;
    red[tid] = d0 * d0 + d1 * d1; __syncthreads();
    for (int s = 128; s > 0; s >>= 1) {
        if (tid < s) red[tid] += red[tid + s];
        __syncthreads();
    }
    float rstd = rsqrtf(red[0] / Dim + 1e-6f);
    x[base + tid]       = d0 * rstd * g[tid]       + bta[tid];
    x[base + tid + 256] = d1 * rstd * g[tid + 256] + bta[tid + 256];
}

// ---- out[t, j] = embed[tok[t], j]*scale + pos[t % SeqLen, j]
__global__ __launch_bounds__(256) void embed_k(
    float* __restrict__ out, const int* __restrict__ tok,
    const float* __restrict__ emb, const float* __restrict__ pos,
    int Dim, int SeqLen, float scale)
{
    int idx = blockIdx.x * 256 + threadIdx.x;
    int t = idx / Dim;
    int j = idx - t * Dim;
    int s = t % SeqLen;
    out[idx] = emb[tok[t] * Dim + j] * scale + pos[s * Dim + j];
}

// ---------------------------------------------------------------------------
static inline void launch_gemm(hipStream_t st,
    const float* A, int lda, int aO, int aI,
    const float* Bt, int bsn, int bsk, int bO, int bI,
    float* C, int ldc, int cO, int cI,
    const float* bias, float scale, int relu,
    int M, int N, int K, int innerCnt, int nbatch)
{
    dim3 grid(N / TN, M / TM, nbatch);
    if (bsk == 1)
        gemm_bf16_wmma<true><<<grid, 256, 0, st>>>(
            A, lda, aO, aI, Bt, bsn, bsk, bO, bI,
            C, ldc, cO, cI, bias, scale, relu, M, N, K, innerCnt);
    else
        gemm_bf16_wmma<false><<<grid, 256, 0, st>>>(
            A, lda, aO, aI, Bt, bsn, bsk, bO, bI,
            C, ldc, cO, cI, bias, scale, relu, M, N, K, innerCnt);
}

extern "C" void kernel_launch(void* const* d_in, const int* in_sizes, int n_in,
                              void* d_out, int out_size, void* d_ws, size_t ws_size,
                              hipStream_t stream)
{
    (void)in_sizes; (void)n_in; (void)out_size; (void)ws_size;

    constexpr int Bb = 4, Ss = 512, Dd = 512, Ff = 2048, Ll = 6, Hh = 8, Vv = 32000;
    constexpr int HDd = Dd / Hh;                 // 64
    constexpr int NT  = Bb * Ss;                 // 2048 tokens (enc == dec)
    constexpr int SD  = Ss * Dd;                 // per-batch token block
    constexpr int SS  = Ss * Ss;                 // per-head score block

    const int*   enc_in     = (const int*)  d_in[0];
    const int*   dec_in     = (const int*)  d_in[1];
    const float* e_embed    = (const float*)d_in[2];
    const float* d_embed    = (const float*)d_in[3];
    const float* pos_enc    = (const float*)d_in[4];
    const float* enc_mha_W  = (const float*)d_in[5];
    const float* enc_mha_b  = (const float*)d_in[6];
    const float* enc_ln_g   = (const float*)d_in[7];
    const float* enc_ln_b   = (const float*)d_in[8];
    const float* enc_ffn_W1 = (const float*)d_in[9];
    const float* enc_ffn_b1 = (const float*)d_in[10];
    const float* enc_ffn_W2 = (const float*)d_in[11];
    const float* enc_ffn_b2 = (const float*)d_in[12];
    const float* dec_mha_W  = (const float*)d_in[13];
    const float* dec_mha_b  = (const float*)d_in[14];
    const float* dec_ln_g   = (const float*)d_in[15];
    const float* dec_ln_b   = (const float*)d_in[16];
    const float* dec_ffn_W1 = (const float*)d_in[17];
    const float* dec_ffn_b1 = (const float*)d_in[18];
    const float* dec_ffn_W2 = (const float*)d_in[19];
    const float* dec_ffn_b2 = (const float*)d_in[20];
    float* out = (float*)d_out;

    // workspace layout (floats): ~80 MB total
    constexpr int ACT = NT * Dd;                 // 1,048,576
    float* ws     = (float*)d_ws;
    float* e_act  = ws;
    float* d_act  = ws + 1 * (size_t)ACT;
    float* q      = ws + 2 * (size_t)ACT;
    float* kk     = ws + 3 * (size_t)ACT;
    float* vv     = ws + 4 * (size_t)ACT;
    float* ao     = ws + 5 * (size_t)ACT;   // attention output (concat heads)
    float* y      = ws + 6 * (size_t)ACT;   // sublayer output
    float* hh     = ws + 7 * (size_t)ACT;   // FFN hidden, NT*F = 4*ACT
    float* sc     = ws + 11 * (size_t)ACT;  // scores, B*H*S*S = 8*ACT

    const float escale = sqrtf((float)Dd);

    // ---- embeddings + positional encodings
    embed_k<<<(NT * Dd) / 256, 256, 0, stream>>>(e_act, enc_in, e_embed, pos_enc, Dd, Ss, escale);
    embed_k<<<(NT * Dd) / 256, 256, 0, stream>>>(d_act, dec_in, d_embed, pos_enc, Dd, Ss, escale);

    // shared attention sub-pipeline
    auto attention = [&](float* xq, float* xkv,
                         const float* W, const float* bb,
                         const int* key_tok, int causal) {
        const int DD = Dd * Dd;
        // Q,K,V projections  [NT,D] x [D,D]
        launch_gemm(stream, xq,  Dd, 0, 0, W + 0 * DD, 1, Dd, 0, 0, q,  Dd, 0, 0, bb + 0 * Dd, 1.f, 0, NT, Dd, Dd, 1, 1);
        launch_gemm(stream, xkv, Dd, 0, 0, W + 1 * DD, 1, Dd, 0, 0, kk, Dd, 0, 0, bb + 1 * Dd, 1.f, 0, NT, Dd, Dd, 1, 1);
        launch_gemm(stream, xkv, Dd, 0, 0, W + 2 * DD, 1, Dd, 0, 0, vv, Dd, 0, 0, bb + 2 * Dd, 1.f, 0, NT, Dd, Dd, 1, 1);
        // scores = Q . K^T / sqrt(HD)   batched over B*H (per-head strided views)
        launch_gemm(stream, q,  Dd, SD, HDd,
                            kk, Dd, 1, SD, HDd,
                            sc, Ss, Hh * SS, SS,
                            nullptr, 0.125f, 0, Ss, Ss, HDd, Hh, Bb * Hh);
        softmax_mask_k<<<Bb * Hh * Ss, 256, 0, stream>>>(sc, key_tok, causal, Ss, Hh * Ss);
        // O = P . V  -> write back into [B,S,H,HD] (== [NT,D]) layout
        launch_gemm(stream, sc, Ss, Hh * SS, SS,
                            vv, 1, Dd, SD, HDd,
                            ao, Dd, SD, HDd,
                            nullptr, 1.f, 0, Ss, HDd, Ss, Hh, Bb * Hh);
        // output projection
        launch_gemm(stream, ao, Dd, 0, 0, W + 3 * DD, 1, Dd, 0, 0, y, Dd, 0, 0, bb + 3 * Dd, 1.f, 0, NT, Dd, Dd, 1, 1);
    };

    // ================= encoder =================
    for (int l = 0; l < Ll; ++l) {
        attention(e_act, e_act,
                  enc_mha_W + (size_t)l * 4 * Dd * Dd,
                  enc_mha_b + (size_t)l * 4 * Dd,
                  enc_in, 0);
        add_ln_k<<<NT, 256, 0, stream>>>(e_act, y,
            enc_ln_g + (size_t)(l * 2 + 0) * Dd, enc_ln_b + (size_t)(l * 2 + 0) * Dd, Dd);
        launch_gemm(stream, e_act, Dd, 0, 0, enc_ffn_W1 + (size_t)l * Dd * Ff, 1, Ff, 0, 0,
                    hh, Ff, 0, 0, enc_ffn_b1 + (size_t)l * Ff, 1.f, 1, NT, Ff, Dd, 1, 1);
        launch_gemm(stream, hh, Ff, 0, 0, enc_ffn_W2 + (size_t)l * Ff * Dd, 1, Dd, 0, 0,
                    y, Dd, 0, 0, enc_ffn_b2 + (size_t)l * Dd, 1.f, 0, NT, Dd, Ff, 1, 1);
        add_ln_k<<<NT, 256, 0, stream>>>(e_act, y,
            enc_ln_g + (size_t)(l * 2 + 1) * Dd, enc_ln_b + (size_t)(l * 2 + 1) * Dd, Dd);
    }

    // ================= decoder =================
    for (int l = 0; l < Ll; ++l) {
        const float* W  = dec_mha_W + (size_t)l * 8 * Dd * Dd;
        const float* bb = dec_mha_b + (size_t)l * 8 * Dd;
        // masked self-attention (causal + dec pad)
        attention(d_act, d_act, W, bb, dec_in, 1);
        add_ln_k<<<NT, 256, 0, stream>>>(d_act, y,
            dec_ln_g + (size_t)(l * 3 + 0) * Dd, dec_ln_b + (size_t)(l * 3 + 0) * Dd, Dd);
        // cross-attention (enc pad on keys)
        attention(d_act, e_act, W + 4 * Dd * Dd, bb + 4 * Dd, enc_in, 0);
        add_ln_k<<<NT, 256, 0, stream>>>(d_act, y,
            dec_ln_g + (size_t)(l * 3 + 1) * Dd, dec_ln_b + (size_t)(l * 3 + 1) * Dd, Dd);
        // FFN
        launch_gemm(stream, d_act, Dd, 0, 0, dec_ffn_W1 + (size_t)l * Dd * Ff, 1, Ff, 0, 0,
                    hh, Ff, 0, 0, dec_ffn_b1 + (size_t)l * Ff, 1.f, 1, NT, Ff, Dd, 1, 1);
        launch_gemm(stream, hh, Ff, 0, 0, dec_ffn_W2 + (size_t)l * Ff * Dd, 1, Dd, 0, 0,
                    y, Dd, 0, 0, dec_ffn_b2 + (size_t)l * Dd, 1.f, 0, NT, Dd, Ff, 1, 1);
        add_ln_k<<<NT, 256, 0, stream>>>(d_act, y,
            dec_ln_g + (size_t)(l * 3 + 2) * Dd, dec_ln_b + (size_t)(l * 3 + 2) * Dd, Dd);
    }

    // ---- logits = d @ d_embed^T : B^T(n,k) = d_embed[n*D + k] (contiguous k)
    launch_gemm(stream, d_act, Dd, 0, 0, d_embed, Dd, 1, 0, 0,
                out, Vv, 0, 0, nullptr, 1.f, 0, NT, Vv, Dd, 1, 1);
}
